// FlashGlm4moeAttention_47356309405777
// MI455X (gfx1250) — compile-verified, matrix operand order checked by reading, not run on hardware
//
#include <hip/hip_runtime.h>
#include <hip/hip_bf16.h>

// ---------------------------------------------------------------------------
// Fused GLM4-MoE attention block for MI455X (gfx1250, wave32, WMMA).
// All matmuls: v_wmma_f32_16x16x32_bf16 (fp32 accumulate).
// GEMMs are LDS-staged (128x128 block tile) with double-buffered
// global_load_async_to_lds_b128 fills (ASYNCcnt) overlapped with WMMA.
// ---------------------------------------------------------------------------

#define T_SEQ 2048
#define HID   4096
#define NH    32
#define NKV   8
#define DH    128
#define QKV_N ((NH + 2 * NKV) * DH) // 6144
#define SCALE 0.08838834764831845f  // 128^-0.5
#define LN_ROPE_BASE 9.210340371976184f // ln(10000)

#define BM 128
#define BN 128
#define BK 32

typedef __bf16 bf16_t;
typedef __attribute__((ext_vector_type(16))) __bf16 v16bf;
typedef __attribute__((ext_vector_type(8)))  __bf16 v8bf;
typedef __attribute__((ext_vector_type(8)))  float  v8f;

union AFrag { v16bf v; v8bf h[2]; };

static __device__ __forceinline__ v8f wmma_bf16(v16bf a, v16bf b, v8f c) {
    return __builtin_amdgcn_wmma_f32_16x16x32_bf16(
        /*neg_a=*/false, a, /*neg_b=*/false, b,
        /*c_mod=*/(short)0, c, /*reuse_a=*/false, /*reuse_b=*/false);
}

// Async 16-byte copy global -> LDS (ASYNCcnt-tracked, no VGPR data path).
static __device__ __forceinline__ void async_ld16(const bf16_t* g, bf16_t* l) {
    const unsigned int  loff = (unsigned int)(unsigned long long)(uintptr_t)l;
    const unsigned long long ga = (unsigned long long)(uintptr_t)g;
    asm volatile("global_load_async_to_lds_b128 %0, %1, off"
                 :: "v"(loff), "v"(ga) : "memory");
}
static __device__ __forceinline__ void wait_async0() {
    asm volatile("s_wait_asynccnt 0x0" ::: "memory");
}

static __device__ __forceinline__ float red_max16(float v) {
#pragma unroll
    for (int m = 8; m >= 1; m >>= 1) v = fmaxf(v, __shfl_xor(v, m, 32));
    return v;
}
static __device__ __forceinline__ float red_sum16(float v) {
#pragma unroll
    for (int m = 8; m >= 1; m >>= 1) v += __shfl_xor(v, m, 32);
    return v;
}

// ---------------------------------------------------------------------------
// fp32 -> bf16 conversion (grid-stride)
// ---------------------------------------------------------------------------
__global__ void f32_to_bf16_kernel(const float* __restrict__ in,
                                   bf16_t* __restrict__ out, int n) {
    for (int i = blockIdx.x * blockDim.x + threadIdx.x; i < n;
         i += gridDim.x * blockDim.x)
        out[i] = (bf16_t)in[i];
}

// ---------------------------------------------------------------------------
// C[M,N] = A[M,K](bf16,rm) * B[N,K]^T(bf16,rm) + bias
// Block = 256 threads (8 waves) computing a 128x128 tile.
// Waves: 4 along M x 2 along N; each wave owns 32x64 (8 WMMAs / K-step).
// A/B K-slices staged in LDS, double-buffered, filled by async-to-LDS.
// ---------------------------------------------------------------------------
__global__ void __launch_bounds__(256)
gemm_lds_bf16_bt_kernel(const bf16_t* __restrict__ A,
                        const bf16_t* __restrict__ B,
                        const float* __restrict__ bias, float* __restrict__ C,
                        int M, int N, int K) {
    __shared__ __align__(16) bf16_t Asm[2][BM][BK]; // 2 x 8 KiB
    __shared__ __align__(16) bf16_t Bsm[2][BN][BK]; // 2 x 8 KiB

    const int tid   = threadIdx.x;
    const int lane  = tid & 31;
    const int wv    = tid >> 5;   // 0..7
    const int wm    = wv & 3;     // wave row (4 along M)
    const int wn    = wv >> 2;    // wave col (2 along N)
    const int tilesN = N / BN;
    const int bm = blockIdx.x / tilesN;
    const int bn = blockIdx.x % tilesN;

    const int lanem = lane & 15;
    const int half  = lane >> 4;
    const int kb    = half * 8;

    // Each K-slice is 128 rows x 32 bf16 = 8 KiB = 512 x 16B chunks per
    // matrix; 256 threads -> 2 chunks of A + 2 chunks of B per thread.
    auto issue_tile = [&](int k0, int s) {
#pragma unroll
        for (int i = 0; i < 2; ++i) {
            const int c    = tid + i * 256;
            const int row  = c >> 2;         // 4 chunks per row
            const int part = (c & 3) * 8;    // bf16 offset within row
            async_ld16(A + (size_t)(bm * BM + row) * K + k0 + part,
                       &Asm[s][row][part]);
        }
#pragma unroll
        for (int i = 0; i < 2; ++i) {
            const int c    = tid + i * 256;
            const int row  = c >> 2;
            const int part = (c & 3) * 8;
            async_ld16(B + (size_t)(bn * BN + row) * K + k0 + part,
                       &Bsm[s][row][part]);
        }
    };

    v8f c[2][4];
#pragma unroll
    for (int i = 0; i < 2; ++i)
#pragma unroll
        for (int j = 0; j < 4; ++j) c[i][j] = {};

    issue_tile(0, 0);
    wait_async0();
    __syncthreads();

    int buf = 0;
    for (int k0 = 0; k0 < K; k0 += BK) {
        const bool more = (k0 + BK) < K;
        if (more) issue_tile(k0 + BK, buf ^ 1); // overlap fill with compute

        AFrag a0, a1;
        a0.h[0] = *(const v8bf*)&Asm[buf][wm * 32 + lanem][kb];
        a0.h[1] = *(const v8bf*)&Asm[buf][wm * 32 + lanem][16 + kb];
        a1.h[0] = *(const v8bf*)&Asm[buf][wm * 32 + 16 + lanem][kb];
        a1.h[1] = *(const v8bf*)&Asm[buf][wm * 32 + 16 + lanem][16 + kb];
#pragma unroll
        for (int j = 0; j < 4; ++j) {
            AFrag bf;
            bf.h[0] = *(const v8bf*)&Bsm[buf][wn * 64 + j * 16 + lanem][kb];
            bf.h[1] = *(const v8bf*)&Bsm[buf][wn * 64 + j * 16 + lanem][16 + kb];
            c[0][j] = wmma_bf16(a0.v, bf.v, c[0][j]);
            c[1][j] = wmma_bf16(a1.v, bf.v, c[1][j]);
        }

        if (more) {
            wait_async0();     // own wave's async fills done (in-order)
            __syncthreads();   // -> whole tile complete, old reads done
        }
        buf ^= 1;
    }

#pragma unroll
    for (int i = 0; i < 2; ++i)
#pragma unroll
        for (int j = 0; j < 4; ++j)
#pragma unroll
            for (int r = 0; r < 8; ++r) {
                const int row = bm * BM + wm * 32 + i * 16 + r + half * 8;
                const int col = bn * BN + wn * 64 + j * 16 + lanem;
                C[(size_t)row * N + col] = c[i][j][r] + (bias ? bias[col] : 0.0f);
            }
}

// ---------------------------------------------------------------------------
// RMSNorm + RoPE + scale + layout change. One wave per (t, slot),
// slot 0..31 = q head, 32..39 = k head, 40..47 = v head.
// Q -> [NH][T][DH] bf16 (pre-scaled by SCALE), K -> [NKV][T][DH] bf16,
// V -> transposed [NKV][DH][T] bf16 (so PV B-fragments load contiguously).
// ---------------------------------------------------------------------------
__global__ void __launch_bounds__(128)
qkv_prep_kernel(const float* __restrict__ qkv,
                const float* __restrict__ qw, const float* __restrict__ kw,
                bf16_t* __restrict__ Qh, bf16_t* __restrict__ Kh,
                bf16_t* __restrict__ Vt) {
    const int lane = threadIdx.x & 31;
    const int wave = blockIdx.x * (blockDim.x >> 5) + (threadIdx.x >> 5);
    const int t    = wave / 48;
    const int slot = wave % 48;
    if (t >= T_SEQ) return;

    if (slot >= 40) { // V: plain copy, transposed
        const int h = slot - 40;
        const float* src = qkv + (size_t)t * QKV_N + (NH + NKV + h) * DH;
#pragma unroll
        for (int i = 0; i < 4; ++i) {
            const int d = lane * 4 + i;
            Vt[((size_t)h * DH + d) * T_SEQ + t] = (bf16_t)src[d];
        }
        return;
    }

    const bool isq = slot < 32;
    const int  h   = isq ? slot : slot - 32;
    const float* src = qkv + (size_t)t * QKV_N + (isq ? h * DH : (NH + h) * DH);

    const int pA = lane, pB = lane + 32; // rotation-pair indices (0..63)
    float a1 = src[pA], a2 = src[pA + 64];
    float b1 = src[pB], b2 = src[pB + 64];

    float ss = a1 * a1 + a2 * a2 + b1 * b1 + b2 * b2;
#pragma unroll
    for (int m = 16; m >= 1; m >>= 1) ss += __shfl_xor(ss, m, 32);
    const float r = rsqrtf(ss * (1.0f / 128.0f) + 1e-6f);

    const float* nw = isq ? qw : kw;
    a1 *= r * nw[pA]; a2 *= r * nw[pA + 64];
    b1 *= r * nw[pB]; b2 *= r * nw[pB + 64];

    const float fA = (float)t * __expf(-((float)(2 * pA) / 128.0f) * LN_ROPE_BASE);
    const float fB = (float)t * __expf(-((float)(2 * pB) / 128.0f) * LN_ROPE_BASE);
    float sA, cA, sB, cB;
    __sincosf(fA, &sA, &cA);
    __sincosf(fB, &sB, &cB);

    const float sc = isq ? SCALE : 1.0f;
    bf16_t* dst = (isq ? Qh : Kh) + ((size_t)h * T_SEQ + t) * DH;
    dst[pA]      = (bf16_t)((a1 * cA - a2 * sA) * sc);
    dst[pA + 64] = (bf16_t)((a2 * cA + a1 * sA) * sc);
    dst[pB]      = (bf16_t)((b1 * cB - b2 * sB) * sc);
    dst[pB + 64] = (bf16_t)((b2 * cB + b1 * sB) * sc);
}

// ---------------------------------------------------------------------------
// Causal flash attention. One wave per (head, 16-query tile).
// Scores via 8 WMMAs per 32-key block; online softmax in fp32; P transposed
// D-layout -> A-layout through a per-wave 1KB LDS tile; PV via 8 WMMAs.
// Output written bf16 into attnB[T][NH*DH] for the final WMMA GEMM.
// ---------------------------------------------------------------------------
__global__ void __launch_bounds__(128)
flash_attn_kernel(const bf16_t* __restrict__ Qh, const bf16_t* __restrict__ Kh,
                  const bf16_t* __restrict__ Vt, bf16_t* __restrict__ attnB) {
    __shared__ __align__(16) bf16_t plds[4][16][32];

    const int lane = threadIdx.x & 31;
    const int w    = threadIdx.x >> 5;
    const int wave = blockIdx.x * 4 + w;
    const int head = wave >> 7;          // T/16 = 128 query tiles per head
    const int q0   = (wave & 127) * 16;
    const int kvh  = head >> 2;          // GQA: 4 q heads per kv head

    const int lanem = lane & 15;
    const int half  = lane >> 4;
    const int kb    = half * 8;

    // Q fragments for the whole D=128 (4 chunks of K=32), kept in registers.
    AFrag qa[4];
    {
        const bf16_t* qbase = Qh + ((size_t)head * T_SEQ + q0 + lanem) * DH;
#pragma unroll
        for (int kk = 0; kk < 4; ++kk) {
            qa[kk].h[0] = *(const v8bf*)(qbase + kk * 32 + kb);
            qa[kk].h[1] = *(const v8bf*)(qbase + kk * 32 + 16 + kb);
        }
    }

    float m8[8], l8[8];
    v8f acc[8];
#pragma unroll
    for (int i = 0; i < 8; ++i) { m8[i] = -INFINITY; l8[i] = 0.0f; acc[i] = {}; }

    for (int j0 = 0; j0 < q0 + 16; j0 += 32) {
        // --- scores: two 16x16 tiles covering keys j0..j0+31 ---
        v8f s0 = {}; v8f s1 = {};
#pragma unroll
        for (int kk = 0; kk < 4; ++kk) {
            AFrag kf0, kf1;
            const bf16_t* kbase = Kh + ((size_t)kvh * T_SEQ + j0 + lanem) * DH + kk * 32;
            kf0.h[0] = *(const v8bf*)(kbase + kb);
            kf0.h[1] = *(const v8bf*)(kbase + 16 + kb);
            kf1.h[0] = *(const v8bf*)(kbase + (size_t)16 * DH + kb);
            kf1.h[1] = *(const v8bf*)(kbase + (size_t)16 * DH + 16 + kb);
            s0 = wmma_bf16(qa[kk].v, kf0.v, s0);
            s1 = wmma_bf16(qa[kk].v, kf1.v, s1);
        }

        // --- causal mask + online softmax (row = r + half*8, col = lanem) ---
#pragma unroll
        for (int r = 0; r < 8; ++r) {
            const int q  = q0 + r + half * 8;
            const int c0 = j0 + lanem;
            const int c1 = j0 + 16 + lanem;
            const float v0 = (c0 <= q) ? s0[r] : -3.0e38f;
            const float v1 = (c1 <= q) ? s1[r] : -3.0e38f;
            const float rm = red_max16(fmaxf(v0, v1));
            const float mn = fmaxf(m8[r], rm);
            const float scl = __expf(m8[r] - mn);
            const float p0 = __expf(v0 - mn);
            const float p1 = __expf(v1 - mn);
            plds[w][r + half * 8][lanem]      = (bf16_t)p0;
            plds[w][r + half * 8][16 + lanem] = (bf16_t)p1;
            l8[r] = l8[r] * scl + red_sum16(p0 + p1);
            m8[r] = mn;
#pragma unroll
            for (int n = 0; n < 8; ++n) acc[n][r] *= scl;
        }

        // DS ops are in-order within a wave; drain stores before reloading.
        asm volatile("s_wait_dscnt 0x0" ::: "memory");

        AFrag pa;
        pa.h[0] = *(const v8bf*)&plds[w][lanem][kb];
        pa.h[1] = *(const v8bf*)&plds[w][lanem][16 + kb];

        // --- O += P * V  (V pre-transposed: rows = head-dim, cols = time) ---
#pragma unroll
        for (int n = 0; n < 8; ++n) {
            AFrag vf;
            const bf16_t* vbase =
                Vt + ((size_t)kvh * DH + n * 16 + lanem) * T_SEQ + j0;
            vf.h[0] = *(const v8bf*)(vbase + kb);
            vf.h[1] = *(const v8bf*)(vbase + 16 + kb);
            acc[n] = wmma_bf16(pa.v, vf.v, acc[n]);
        }
    }

    // --- normalize and store bf16 output row-major [T][NH*DH] ---
#pragma unroll
    for (int n = 0; n < 8; ++n) {
#pragma unroll
        for (int r = 0; r < 8; ++r) {
            const int q = q0 + r + half * 8;
            const float o = acc[n][r] / l8[r];
            attnB[(size_t)q * (NH * DH) + head * DH + n * 16 + lanem] = (bf16_t)o;
        }
    }
}

// ---------------------------------------------------------------------------
// Host-side orchestration (graph-capture safe: only kernel launches on stream)
// ---------------------------------------------------------------------------
extern "C" void kernel_launch(void* const* d_in, const int* in_sizes, int n_in,
                              void* d_out, int out_size, void* d_ws, size_t ws_size,
                              hipStream_t stream) {
    const float* x        = (const float*)d_in[0];
    const float* w_qkv    = (const float*)d_in[1];
    const float* b_qkv    = (const float*)d_in[2];
    const float* q_norm_w = (const float*)d_in[3];
    const float* k_norm_w = (const float*)d_in[4];
    const float* w_o      = (const float*)d_in[5];
    float* out = (float*)d_out;

    // Workspace layout (bytes). attnB aliases xb (xb is dead after QKV GEMM).
    char* ws = (char*)d_ws;
    const size_t sz_xb    = (size_t)2 * T_SEQ * HID;        // 16 MiB
    const size_t sz_wqkvb = (size_t)2 * QKV_N * HID;        // 48 MiB
    const size_t sz_wob   = (size_t)2 * HID * HID;          // 32 MiB
    const size_t sz_qkvf  = (size_t)4 * T_SEQ * QKV_N;      // 48 MiB
    const size_t sz_qh    = (size_t)2 * NH * T_SEQ * DH;    // 16 MiB
    const size_t sz_kh    = (size_t)2 * NKV * T_SEQ * DH;   //  4 MiB
    size_t off = 0;
    bf16_t* xb    = (bf16_t*)(ws + off); off += sz_xb;
    bf16_t* wqkvb = (bf16_t*)(ws + off); off += sz_wqkvb;
    bf16_t* wob   = (bf16_t*)(ws + off); off += sz_wob;
    float*  qkvf  = (float*) (ws + off); off += sz_qkvf;
    bf16_t* Qh    = (bf16_t*)(ws + off); off += sz_qh;
    bf16_t* Kh    = (bf16_t*)(ws + off); off += sz_kh;
    bf16_t* Vt    = (bf16_t*)(ws + off); off += sz_kh;
    bf16_t* attnB = xb; // alias: reuse after xb consumed

    // 1) fp32 -> bf16 converts
    f32_to_bf16_kernel<<<2048, 256, 0, stream>>>(x, xb, T_SEQ * HID);
    f32_to_bf16_kernel<<<4096, 256, 0, stream>>>(w_qkv, wqkvb, QKV_N * HID);
    f32_to_bf16_kernel<<<4096, 256, 0, stream>>>(w_o, wob, HID * HID);

    // 2) qkv = x @ w_qkv^T + b : (2048 x 6144), 16 x 48 = 768 block tiles
    gemm_lds_bf16_bt_kernel<<<(T_SEQ / BM) * (QKV_N / BN), 256, 0, stream>>>(
        xb, wqkvb, b_qkv, qkvf, T_SEQ, QKV_N, HID);

    // 3) RMSNorm + RoPE + layout: 2048 * 48 waves
    qkv_prep_kernel<<<(T_SEQ * 48) / 4, 128, 0, stream>>>(
        qkvf, q_norm_w, k_norm_w, Qh, Kh, Vt);

    // 4) causal flash attention: 32 heads * 128 query tiles = 4096 waves
    flash_attn_kernel<<<(NH * (T_SEQ / 16)) / 4, 128, 0, stream>>>(
        Qh, Kh, Vt, attnB);

    // 5) out = attn @ w_o^T : (2048 x 4096), 16 x 32 = 512 block tiles
    gemm_lds_bf16_bt_kernel<<<(T_SEQ / BM) * (HID / BN), 256, 0, stream>>>(
        attnB, wob, nullptr, out, T_SEQ, HID, HID);
}
